// HyperNetworkNFN_78194174591250
// MI455X (gfx1250) — matrix operand order repaired and checked by reading, not compiled, not run on hardware
//
#include <hip/hip_runtime.h>

// ============================================================================
// HyperNetworkNFN on MI455X (gfx1250).
// NFN layers collapsed to GEMMs: per (batch,n1): Y = BigM[4co,4ci] @ X + bias(b)
// BigM shared across batch -> bf16 WMMA GEMMs (fp32 accumulate).
// LDS staging of the B operand uses the Tensor Data Mover when the builtin is
// available (D# built per ISA 8.3/8.4; 5-arg on clang-22 / ROCm 7.2, 6-arg on
// clang-23 / amdgpu-toolchain), else a manual vector copy.
// ============================================================================

typedef __bf16 bf16;
typedef __attribute__((ext_vector_type(16))) __bf16 v16bf;
typedef __attribute__((ext_vector_type(8)))  __bf16 v8bf;
typedef __attribute__((ext_vector_type(8)))  float  v8f;

#define N1D   544
#define BATCH 512
#define FLATD 2177
#define NCOLS (BATCH * N1D)   // 278528
#define COMBP 2208            // FLATD padded to multiple of 32

#if defined(__gfx1250__) && \
    __has_builtin(__builtin_amdgcn_tensor_load_to_lds) && \
    __has_builtin(__builtin_amdgcn_s_wait_tensorcnt)
  #if (__clang_major__ == 22)
    #define NFN_TDM_ARITY 5
    #pragma message("NFN: TDM path enabled (5-arg, clang-22)")
  #else
    #define NFN_TDM_ARITY 6
    #pragma message("NFN: TDM path enabled (6-arg, clang>=23)")
  #endif
#else
  #define NFN_TDM_ARITY 0
  #pragma message("NFN: TDM path disabled (manual LDS copy)")
#endif

// ---------------------------------------------------------------------------
__global__ void k_pack(const float* __restrict__ x, bf16* __restrict__ X0,
                       float* __restrict__ b1a) {
  int b = blockIdx.x;
  const float* xb = x + (size_t)b * (2 * FLATD);
  for (int n = threadIdx.x; n < N1D; n += blockDim.x) {
    bf16* row = X0 + ((size_t)b * N1D + n) * 32;
    row[0] = (bf16)xb[2 * n];         row[1] = (bf16)xb[FLATD + 2 * n];
    row[2] = (bf16)xb[2 * n + 1];     row[3] = (bf16)xb[FLATD + 2 * n + 1];
    row[4] = (bf16)xb[1088 + n];      row[5] = (bf16)xb[FLATD + 1088 + n];
    row[6] = (bf16)xb[1632 + n];      row[7] = (bf16)xb[FLATD + 1632 + n];
#pragma unroll
    for (int i = 8; i < 32; ++i) row[i] = (bf16)0.0f;
  }
  for (int c = threadIdx.x; c < 64; c += blockDim.x) {
    float v = 0.0f;
    if (c == 0) v = xb[2176];
    else if (c == 1) v = xb[FLATD + 2176];
    b1a[b * 64 + c] = v;
  }
}

// Build BigM (bf16 row-major [MP][KP]); row m = oblk*cout+c, col q = iblk*cin+ci
__global__ void k_bigm(const float* __restrict__ wid0, const float* __restrict__ wrow0,
                       const float* __restrict__ wcn0, const float* __restrict__ wbj0,
                       const float* __restrict__ bid0, const float* __restrict__ brow0,
                       const float* __restrict__ bcn0,
                       const float* __restrict__ wid1, const float* __restrict__ wcol1,
                       const float* __restrict__ wrp1, const float* __restrict__ wbp1,
                       bf16* __restrict__ A, int cin, int cout, int KP, int MP) {
  int idx = blockIdx.x * blockDim.x + threadIdx.x;
  if (idx >= MP * KP) return;
  int m = idx / KP, q = idx % KP;
  float v = 0.0f;
  if (m < 4 * cout && q < 4 * cin) {
    int ob = m / cout, c = m % cout;
    int ib = q / cin, ci = q % cin;
    int e = c * cin + ci;
    if (ob == 0)      v = (ib == 0) ? wid0[e] + 0.5f * wrow0[e]
                        : (ib == 1) ? 0.5f * wrow0[e]
                        : (ib == 2) ? wbj0[e] : wcn0[e];
    else if (ob == 1) v = (ib == 0) ? 0.5f * wrow0[e]
                        : (ib == 1) ? wid0[e] + 0.5f * wrow0[e]
                        : (ib == 2) ? wbj0[e] : wcn0[e];
    else if (ob == 2) v = (ib <= 1) ? 0.5f * brow0[e]
                        : (ib == 2) ? bid0[e] : bcn0[e];
    else              v = (ib <= 1) ? 0.5f * wrp1[e]
                        : (ib == 2) ? wbp1[e] : (wid1[e] + wcol1[e]);
  }
  A[(size_t)m * KP + q] = (bf16)v;
}

__global__ void k_cvt(const float* __restrict__ W, bf16* __restrict__ A,
                      int M, int K, int KP, int total) {
  int idx = blockIdx.x * blockDim.x + threadIdx.x;
  if (idx >= total) return;
  int m = idx / KP, q = idx % KP;
  float v = (m < M && q < K) ? W[(size_t)m * K + q] : 0.0f;
  A[idx] = (bf16)v;
}

__global__ void k_zero_bf16(bf16* __restrict__ p, int n) {
  int i = blockIdx.x * blockDim.x + threadIdx.x;
  if (i < n) p[i] = (bf16)0.0f;
}

__global__ void k_padbias(const float* __restrict__ src, float* __restrict__ dst,
                          int n, int np) {
  int i = blockIdx.x * blockDim.x + threadIdx.x;
  if (i < np) dst[i] = (i < n) ? src[i] : 0.0f;
}

// Per-batch row means of X [b][n][r] over n -> stats[b][256]
__global__ void k_stats(const bf16* __restrict__ X, float* __restrict__ stats, int KP) {
  int b = blockIdx.x;
  int r = threadIdx.x;                       // 256 threads
  if (r >= KP) { stats[b * 256 + r] = 0.0f; return; }
  const bf16* base = X + (size_t)b * N1D * KP + r;
  float s = 0.0f;
  for (int n = 0; n < N1D; ++n) {
    __builtin_prefetch(base + (size_t)(n + 16) * KP, 0, 0);   // global_prefetch_b8
    s += (float)base[(size_t)n * KP];
  }
  stats[b * 256 + r] = s * (1.0f / 544.0f);
}

// Per-batch bias vectors for one NFN layer + b1 chain.
__global__ void k_bias(const float* __restrict__ stats, const float* __restrict__ b1cur,
                       const float* __restrict__ wcol,  const float* __restrict__ winv0,
                       const float* __restrict__ wbias0,const float* __restrict__ ein,
                       const float* __restrict__ binv0, const float* __restrict__ bbias0,
                       const float* __restrict__ wrow1, const float* __restrict__ wbj1,
                       const float* __restrict__ winv1, const float* __restrict__ wbias1,
                       const float* __restrict__ eout,
                       const float* __restrict__ bid1,  const float* __restrict__ brow1,
                       const float* __restrict__ binv1, const float* __restrict__ bbias1,
                       const float* __restrict__ ebout,
                       float* __restrict__ biasv, float* __restrict__ b1next,
                       bf16* __restrict__ comb,
                       int cin, int cout, int relu_b1, int is_last) {
  int b = blockIdx.x;
  int c = threadIdx.x;
  if (c >= cout) return;
  const float* st = stats + b * 256;
  const float* b1 = b1cur + b * 64;
  const int s4 = 4 * cin;
  float d_col0 = 0, d_col1 = 0, d_row1 = 0, d_bj1 = 0, d_id1 = 0, d_brow1 = 0;
  float iv_w0 = 0, iv_b0 = 0, iv_w1 = 0, iv_b1 = 0;
  for (int ci = 0; ci < cin; ++ci) {
    float su0 = st[ci], su1 = st[cin + ci], sv = st[2 * cin + ci], sw = st[3 * cin + ci];
    float b1v = b1[ci];
    d_col0  += wcol[c * cin + ci] * su0;
    d_col1  += wcol[c * cin + ci] * su1;
    d_row1  += wrow1[c * cin + ci] * sw;
    d_bj1   += wbj1[c * cin + ci] * b1v;
    d_id1   += bid1[c * cin + ci] * b1v;
    d_brow1 += brow1[c * cin + ci] * sw;
    float i0 = 0.5f * (su0 + su1), i1 = sw, i2 = sv, i3 = b1v;
    iv_w0 += winv0[c*s4+ci]*i0 + winv0[c*s4+cin+ci]*i1 + winv0[c*s4+2*cin+ci]*i2 + winv0[c*s4+3*cin+ci]*i3;
    iv_b0 += binv0[c*s4+ci]*i0 + binv0[c*s4+cin+ci]*i1 + binv0[c*s4+2*cin+ci]*i2 + binv0[c*s4+3*cin+ci]*i3;
    iv_w1 += winv1[c*s4+ci]*i0 + winv1[c*s4+cin+ci]*i1 + winv1[c*s4+2*cin+ci]*i2 + winv1[c*s4+3*cin+ci]*i3;
    iv_b1 += binv1[c*s4+ci]*i0 + binv1[c*s4+cin+ci]*i1 + binv1[c*s4+2*cin+ci]*i2 + binv1[c*s4+3*cin+ci]*i3;
  }
  biasv[b * 256 + 0 * cout + c] = d_col0 + iv_w0 + wbias0[c] + ein[c * 2 + 0];
  biasv[b * 256 + 1 * cout + c] = d_col1 + iv_w0 + wbias0[c] + ein[c * 2 + 1];
  biasv[b * 256 + 2 * cout + c] = iv_b0 + bbias0[c];
  biasv[b * 256 + 3 * cout + c] = d_row1 + d_bj1 + iv_w1 + wbias1[c] + eout[c];
  float b1n = d_id1 + d_brow1 + iv_b1 + bbias1[c] + ebout[c];
  if (relu_b1) b1n = b1n > 0.0f ? b1n : 0.0f;
  b1next[b * 64 + c] = b1n;
  if (is_last && c == 0) comb[(size_t)b * COMBP + 2176] = (bf16)b1n;
}

// ---------------------------------------------------------------------------
__device__ __forceinline__ v16bf load_frag(const bf16* p0, const bf16* p1) {
  v8bf a0 = *(const v8bf*)p0;
  v8bf a1 = *(const v8bf*)p1;
  v16bf r;
#pragma unroll
  for (int j = 0; j < 8; ++j) { r[j] = a0[j]; r[j + 8] = a1[j]; }
  return r;
}

#if NFN_TDM_ARITY
// One TDM descriptor: contiguous 1-D copy of nbytes (multiple of 8) from
// global `ga` into LDS offset `lds_base`. Issued by a single wave.
__device__ __forceinline__ void tdm_copy_1d(unsigned long long ga,
                                            unsigned lds_base, unsigned nbytes) {
  typedef __attribute__((ext_vector_type(4))) unsigned int v4u;
  typedef __attribute__((ext_vector_type(8))) int v8i;
  typedef __attribute__((ext_vector_type(4))) int v4i;
  const unsigned elems = nbytes >> 3;            // 8-byte units (<= 8192 here)
  v4u g0;                                        // count=1 | lds | gaddr | type=2
  g0[0] = 1u;
  g0[1] = lds_base;
  g0[2] = (unsigned)ga;
  g0[3] = ((unsigned)(ga >> 32) & 0x01FFFFFFu) | 0x80000000u;
  v8i g1;                                        // data_size=8B, 1-D tile
  g1[0] = (int)(3u << 16);
  g1[1] = (int)(elems << 16);                    // tensor_dim0[15:0]
  g1[2] = (int)(1u << 16);                       // tensor_dim1 = 1
  g1[3] = (int)(elems << 16);                    // tile_dim0
  g1[4] = 1;                                     // tile_dim1 = 1
  g1[5] = (int)elems;                            // tensor_dim0_stride
  g1[6] = 0; g1[7] = 0;
  v4i gz4 = {0, 0, 0, 0};
#if (NFN_TDM_ARITY == 5)
  __builtin_amdgcn_tensor_load_to_lds(g0, g1, gz4, gz4, 0);
#else
  v8i gz8 = {0, 0, 0, 0, 0, 0, 0, 0};
  __builtin_amdgcn_tensor_load_to_lds(g0, g1, gz4, gz4, gz8, 0);
#endif
  __builtin_amdgcn_s_wait_tensorcnt(0);
}
#endif

// ---------------------------------------------------------------------------
// Generic WMMA GEMM: Out = act(A[MP][KP] @ X[ncols][KP]^T + bias)
// NTHREADS/32 waves per block; block covers 16*NREP columns; each wave owns a
// 16-row M tile (WON mode: all waves share M tile 0 and split the N reps).
// A-fragments preloaded to registers when KP<=256.
// STOREMODE: 0 bf16 n-major [ncols][MOUT]; 1 comb scatter; 2 fp32 [ncols][FLATD]
// BIASMODE : 0 per-batch biasv[b*256+m]; 1 per-m bias[m]  (vector v8f load)
// ---------------------------------------------------------------------------
template <int KP, int MP, int MVALID, int MOUT, bool RELU, int BIASMODE,
          int STOREMODE, bool STAGE, int NREP, bool WON, int NTHREADS>
__global__ __launch_bounds__(NTHREADS) void k_gemm(const bf16* __restrict__ A,
                                                   const bf16* __restrict__ X,
                                                   const float* __restrict__ bias,
                                                   void* __restrict__ out) {
  extern __shared__ char smem_raw[];
  bf16* sX = (bf16*)smem_raw;                       // [16*NREP][KP]

  constexpr int NWAVES = NTHREADS / 32;
  const int tid     = threadIdx.x;
  const int lane    = tid & 31;
  const int wave    = tid >> 5;
  const int colbase = blockIdx.x * (16 * NREP);
  const int mtile   = WON ? 0 : (blockIdx.y * NWAVES + wave);
  const int mbase   = mtile * 16;

  if (STAGE) {
    const unsigned nbytes = 16u * NREP * KP * 2u;   // contiguous global chunk
#if NFN_TDM_ARITY
    if (wave == 0) {
      unsigned long long ga =
          (unsigned long long)(const char*)(X + (size_t)colbase * KP);
      unsigned lds_base =
          (unsigned)(unsigned long long)(__attribute__((address_space(3))) char*)smem_raw;
      tdm_copy_1d(ga, lds_base, nbytes);
    }
#else
    {
      const uint4* src = (const uint4*)(X + (size_t)colbase * KP);
      uint4* dst = (uint4*)sX;
      const int nchunks = (int)(nbytes / 16u);
      for (int i = tid; i < nchunks; i += NTHREADS) dst[i] = src[i];
    }
#endif
    __syncthreads();
  }

  if (!WON && mbase >= MP) return;

  const int hh  = lane >> 4;
  const int l15 = lane & 15;
  const bf16* arow = A + (size_t)(mbase + l15) * KP + hh * 8;

  constexpr int KT = KP / 32;
  constexpr bool AHOLD = (KP <= 256);
  v16bf afr[AHOLD ? KT : 1];
  if (AHOLD) {
#pragma unroll
    for (int kt = 0; kt < KT; ++kt)
      afr[kt] = load_frag(arow + kt * 32, arow + kt * 32 + 16);
  }

  for (int rep = (WON ? wave : 0); rep < NREP; rep += (WON ? NWAVES : 1)) {
    const bf16* bbase = STAGE ? sX : (X + (size_t)colbase * KP);
    const bf16* brow = bbase + (size_t)(rep * 16 + l15) * KP + hh * 8;

    v8f acc = {};
    for (int kt = 0; kt < KT; ++kt) {
      v16bf af = AHOLD ? afr[kt]
                       : load_frag(arow + kt * 32, arow + kt * 32 + 16);
      v16bf bfv = load_frag(brow + kt * 32, brow + kt * 32 + 16);
      acc = __builtin_amdgcn_wmma_f32_16x16x32_bf16(false, af, false, bfv,
                                                    (short)0, acc, false, false);
    }

    // D layout: lane holds column n=lane&15; VGPR i -> row m = i + 8*(lane>=16)
    const int n    = colbase + rep * 16 + l15;
    const int mo   = mbase + hh * 8;
    const int bidx = n / N1D;
    const float* bp = (BIASMODE == 0) ? (bias + (size_t)bidx * 256) : bias;
    v8f bv = *(const v8f*)(bp + mo);                // 32B aligned vector load

    float vals[8];
#pragma unroll
    for (int i = 0; i < 8; ++i) {
      float v = acc[i] + bv[i];
      if (RELU) v = v > 0.0f ? v : 0.0f;
      vals[i] = v;
    }

    if (STOREMODE == 0) {
      bf16* O = (bf16*)out;
      v8bf ov;
#pragma unroll
      for (int i = 0; i < 8; ++i) ov[i] = (bf16)vals[i];
      *(v8bf*)(O + (size_t)n * MOUT + mo) = ov;
    } else if (STOREMODE == 1) {
      bf16* O = (bf16*)out;                        // comb [BATCH][COMBP]
      const int nn = n - bidx * N1D;
#pragma unroll
      for (int i = 0; i < 8; ++i) {
        int m = mo + i;
        if (m < 4) {
          int col = (m == 0) ? 2 * nn
                  : (m == 1) ? (2 * nn + 1)
                  : (m == 2) ? (1632 + nn) : (1088 + nn);
          O[(size_t)bidx * COMBP + col] = (bf16)vals[i];
        }
      }
    } else {
      float* O = (float*)out;                      // [ncols][FLATD]
#pragma unroll
      for (int i = 0; i < 8; ++i) {
        int m = mo + i;
        if (m < MVALID) O[(size_t)n * FLATD + m] = vals[i];
      }
    }
  }
}

// ---------------------------------------------------------------------------
extern "C" void kernel_launch(void* const* d_in, const int* in_sizes, int n_in,
                              void* d_out, int out_size, void* d_ws, size_t ws_size,
                              hipStream_t stream) {
  (void)in_sizes; (void)n_in; (void)out_size; (void)ws_size;
  // Input order: x, p1[27], p2[27], p3[27], fc1_w, fc1_b, fc2_w, fc2_b
  // (dicts flattened in insertion order).
  const float* x = (const float*)d_in[0];
  auto P = [&](int base, int i) { return (const float*)d_in[base + i]; };
  const int P1 = 1, P2 = 28, P3 = 55;
  const float* fc1_w = (const float*)d_in[82];
  const float* fc1_b = (const float*)d_in[83];
  const float* fc2_w = (const float*)d_in[84];
  const float* fc2_b = (const float*)d_in[85];

  char* ws = (char*)d_ws;
  size_t off = 0;
  auto alloc = [&](size_t bytes) -> void* {
    void* p = ws + off;
    off = (off + bytes + 255) & ~(size_t)255;
    return p;
  };
  bf16* A1   = (bf16*)alloc((size_t)256 * 32 * 2);
  bf16* A2   = (bf16*)alloc((size_t)256 * 256 * 2);
  bf16* A3   = (bf16*)alloc((size_t)16 * 256 * 2);
  bf16* Afc1 = (bf16*)alloc((size_t)512 * COMBP * 2);
  bf16* Afc2 = (bf16*)alloc((size_t)2192 * 512 * 2);
  bf16* X0   = (bf16*)alloc((size_t)NCOLS * 32 * 2);
  bf16* Xa   = (bf16*)alloc((size_t)NCOLS * 256 * 2);
  bf16* Xb   = (bf16*)alloc((size_t)NCOLS * 256 * 2);
  bf16* comb = (bf16*)alloc((size_t)BATCH * COMBP * 2);
  bf16* h    = (bf16*)alloc((size_t)512 * 512 * 2);
  float* stats   = (float*)alloc((size_t)BATCH * 256 * 4);
  float* biasv   = (float*)alloc((size_t)BATCH * 256 * 4);
  float* b1a     = (float*)alloc((size_t)BATCH * 64 * 4);
  float* b1b     = (float*)alloc((size_t)BATCH * 64 * 4);
  float* fc2bpad = (float*)alloc((size_t)2192 * 4);

  // ---- parameter preprocessing ----
  k_bigm<<<(256 * 32 + 255) / 256, 256, 0, stream>>>(
      P(P1,0), P(P1,1), P(P1,2), P(P1,3), P(P1,11), P(P1,12), P(P1,13),
      P(P1,5), P(P1,8), P(P1,9), P(P1,10), A1, 2, 64, 32, 256);
  k_bigm<<<(256 * 256 + 255) / 256, 256, 0, stream>>>(
      P(P2,0), P(P2,1), P(P2,2), P(P2,3), P(P2,11), P(P2,12), P(P2,13),
      P(P2,5), P(P2,8), P(P2,9), P(P2,10), A2, 64, 64, 256, 256);
  k_bigm<<<(16 * 256 + 255) / 256, 256, 0, stream>>>(
      P(P3,0), P(P3,1), P(P3,2), P(P3,3), P(P3,11), P(P3,12), P(P3,13),
      P(P3,5), P(P3,8), P(P3,9), P(P3,10), A3, 64, 1, 256, 16);
  k_cvt<<<(512 * COMBP + 255) / 256, 256, 0, stream>>>(fc1_w, Afc1, 512, FLATD, COMBP, 512 * COMBP);
  k_cvt<<<(2192 * 512 + 255) / 256, 256, 0, stream>>>(fc2_w, Afc2, FLATD, 512, 512, 2192 * 512);
  k_padbias<<<(2192 + 255) / 256, 256, 0, stream>>>(fc2_b, fc2bpad, FLATD, 2192);
  k_zero_bf16<<<(BATCH * COMBP + 255) / 256, 256, 0, stream>>>(comb, BATCH * COMBP);
  k_pack<<<BATCH, 256, 0, stream>>>(x, X0, b1a);

  // ---- NFN layer 1 (cin=2 -> cout=64), ReLU ----
  k_stats<<<BATCH, 256, 0, stream>>>(X0, stats, 32);
  k_bias<<<BATCH, 64, 0, stream>>>(stats, b1a,
      P(P1,4), P(P1,16), P(P1,20), P(P1,24), P(P1,18), P(P1,22),
      P(P1,6), P(P1,7), P(P1,17), P(P1,21), P(P1,25),
      P(P1,14), P(P1,15), P(P1,19), P(P1,23), P(P1,26),
      biasv, b1b, comb, 2, 64, 1, 0);
  k_gemm<32, 256, 256, 256, true, 0, 0, true, 8, false, 512>
      <<<dim3(NCOLS / 128, 1), 512, 8 * 16 * 32 * 2, stream>>>(A1, X0, biasv, Xa);

  // ---- NFN layer 2 (64 -> 64), ReLU ----
  k_stats<<<BATCH, 256, 0, stream>>>(Xa, stats, 256);
  k_bias<<<BATCH, 64, 0, stream>>>(stats, b1b,
      P(P2,4), P(P2,16), P(P2,20), P(P2,24), P(P2,18), P(P2,22),
      P(P2,6), P(P2,7), P(P2,17), P(P2,21), P(P2,25),
      P(P2,14), P(P2,15), P(P2,19), P(P2,23), P(P2,26),
      biasv, b1a, comb, 64, 64, 1, 0);
  k_gemm<256, 256, 256, 256, true, 0, 0, true, 4, false, 512>
      <<<dim3(NCOLS / 64, 1), 512, 4 * 16 * 256 * 2, stream>>>(A2, Xa, biasv, Xb);

  // ---- NFN layer 3 (64 -> 1), no ReLU, scatter into comb (waves-on-N) ----
  k_stats<<<BATCH, 256, 0, stream>>>(Xb, stats, 256);
  k_bias<<<BATCH, 64, 0, stream>>>(stats, b1a,
      P(P3,4), P(P3,16), P(P3,20), P(P3,24), P(P3,18), P(P3,22),
      P(P3,6), P(P3,7), P(P3,17), P(P3,21), P(P3,25),
      P(P3,14), P(P3,15), P(P3,19), P(P3,23), P(P3,26),
      biasv, b1b, comb, 64, 1, 0, 1);
  k_gemm<256, 16, 16, 1, false, 0, 1, true, 8, true, 256>
      <<<dim3(NCOLS / 128, 1), 256, 8 * 16 * 256 * 2, stream>>>(A3, Xb, biasv, comb);

  // ---- fc1: h = relu(fc1_w @ comb^T + fc1_b)  [M=512, K=2208, N=512] ----
  k_gemm<COMBP, 512, 512, 512, true, 1, 0, false, 2, false, 512>
      <<<dim3(BATCH / 32, 2), 512, 0, stream>>>(Afc1, comb, fc1_b, h);

  // ---- fc2: out = fc2_w @ h^T + fc2_b  [M=2192(pad 2177), K=512, N=512] ----
  k_gemm<512, 2192, FLATD, 1, false, 1, 2, true, 2, false, 512>
      <<<dim3(BATCH / 32, 9), 512, 2 * 16 * 512 * 2, stream>>>(Afc2, h, fc2bpad, d_out);
}
